// SelfAttention_30477087932901
// MI455X (gfx1250) — matrix operand order repaired and checked by reading, not compiled, not run on hardware
//
#include <hip/hip_runtime.h>
#include <hip/hip_bf16.h>

// Problem constants (fixed by the reference).
#define B_  4
#define T_  2048
#define C_  1024
#define BT_ (B_ * T_)

typedef __attribute__((ext_vector_type(16))) _Float16 v16h;
typedef __attribute__((ext_vector_type(8)))  _Float16 v8h;
typedef __attribute__((ext_vector_type(8)))  float    v8f;

// GEMM tiling: 128(M) x 64(N) block tile, K-step 32, 256 threads = 8 waves,
// waves arranged 4(M) x 2(N), each wave owns a 32x32 output patch = 2x2 WMMA tiles.
#define TM 128
#define TN 64
#define TK 32
#define LDSS 40   // LDS row stride in halves: 80 bytes, keeps 16B alignment

// ---- fragment loaders (per CDNA5 ISA 7.12.2 layouts) ---------------------

// A-matrix 16x32 f16: lane L<16 holds row M=L, halves [0..7]=K0..7, [8..15]=K16..23
// lane L>=16 holds row M=L-16, halves [0..7]=K8..15, [8..15]=K24..31
__device__ __forceinline__ v16h load_a_frag(const _Float16* sA, int rowBase, int laneLo, int sel) {
    int base = (rowBase + laneLo) * LDSS + sel * 8;
    v8h lo = *(const v8h*)(sA + base);
    v8h hi = *(const v8h*)(sA + base + 16);
    v16h a;
#pragma unroll
    for (int i = 0; i < 8; ++i) { a[i] = lo[i]; a[i + 8] = hi[i]; }
    return a;
}

// B-matrix 32x16 f16: lane L holds column N=L%16; halves [0..7]=K 8*(L/16)+0..7, [8..15]=K 16+8*(L/16)+0..7.
// We always feed B from a row-major [N][K] tile (W rows / K rows / Vt rows),
// so each lane reads two 8-half runs.
__device__ __forceinline__ v16h load_b_frag(const _Float16* sB, int rowBase, int laneLo, int sel) {
    int base = (rowBase + laneLo) * LDSS + sel * 8;
    v8h lo = *(const v8h*)(sB + base);
    v8h hi = *(const v8h*)(sB + base + 16);
    v16h b;
#pragma unroll
    for (int i = 0; i < 8; ++i) { b[i] = lo[i]; b[i + 8] = hi[i]; }
    return b;
}

__device__ __forceinline__ void wave_mma_step(const _Float16* sA, const _Float16* sB,
                                              int m0, int n0, int laneLo, int sel,
                                              v8f acc[2][2]) {
    v16h a0 = load_a_frag(sA, m0,      laneLo, sel);
    v16h a1 = load_a_frag(sA, m0 + 16, laneLo, sel);
    v16h b0 = load_b_frag(sB, n0,      laneLo, sel);
    v16h b1 = load_b_frag(sB, n0 + 16, laneLo, sel);
    acc[0][0] = __builtin_amdgcn_wmma_f32_16x16x32_f16(false, a0, false, b0, (short)0, acc[0][0], false, false);
    acc[0][1] = __builtin_amdgcn_wmma_f32_16x16x32_f16(false, a0, false, b1, (short)0, acc[0][1], false, false);
    acc[1][0] = __builtin_amdgcn_wmma_f32_16x16x32_f16(false, a1, false, b0, (short)0, acc[1][0], false, false);
    acc[1][1] = __builtin_amdgcn_wmma_f32_16x16x32_f16(false, a1, false, b1, (short)0, acc[1][1], false, false);
}

// ---- kernel 1: QKV projection  y = x @ W^T -------------------------------
// z=0 -> Q (Wq), z=1 -> K (Wk), z=2 -> V stored transposed Vt[b][c][t].
__global__ __launch_bounds__(256) void qkv_kernel(
    const float* __restrict__ x,
    const float* __restrict__ Wq, const float* __restrict__ Wk, const float* __restrict__ Wv,
    _Float16* __restrict__ Q, _Float16* __restrict__ K, _Float16* __restrict__ Vt)
{
    const int which = blockIdx.z;
    const float* W = (which == 0) ? Wq : (which == 1) ? Wk : Wv;
    const int mBase = blockIdx.y * TM;   // row in flattened [B*T]
    const int nBase = blockIdx.x * TN;   // output feature

    __shared__ __align__(16) _Float16 sA[TM * LDSS];
    __shared__ __align__(16) _Float16 sB[TN * LDSS];

    const int tid = threadIdx.x;
    const int wave = tid >> 5, lane = tid & 31, laneLo = lane & 15, sel = lane >> 4;
    const int m0 = (wave >> 1) * 32, n0 = (wave & 1) * 32;

    const v8f vz = {0, 0, 0, 0, 0, 0, 0, 0};
    v8f acc[2][2] = {{vz, vz}, {vz, vz}};

    for (int kk = 0; kk < C_; kk += TK) {
        __syncthreads();
        // stage x tile (128 x 32 f32 -> f16)
#pragma unroll
        for (int p = 0; p < 4; ++p) {
            int row = p * 32 + (tid >> 3);
            int col = (tid & 7) * 4;
            float4 f = *(const float4*)(x + (size_t)(mBase + row) * C_ + kk + col);
            union { _Float16 h[4]; float2 f2; } pk;
            pk.h[0] = (_Float16)f.x; pk.h[1] = (_Float16)f.y;
            pk.h[2] = (_Float16)f.z; pk.h[3] = (_Float16)f.w;
            *(float2*)(sA + row * LDSS + col) = pk.f2;
        }
        // stage W tile (64 x 32 f32 -> f16)
#pragma unroll
        for (int p = 0; p < 2; ++p) {
            int row = p * 32 + (tid >> 3);
            int col = (tid & 7) * 4;
            float4 f = *(const float4*)(W + (size_t)(nBase + row) * C_ + kk + col);
            union { _Float16 h[4]; float2 f2; } pk;
            pk.h[0] = (_Float16)f.x; pk.h[1] = (_Float16)f.y;
            pk.h[2] = (_Float16)f.z; pk.h[3] = (_Float16)f.w;
            *(float2*)(sB + row * LDSS + col) = pk.f2;
        }
        __syncthreads();
        wave_mma_step(sA, sB, m0, n0, laneLo, sel, acc);
    }

    if (which == 2) {
        // transposed store: each lane owns 8 contiguous t values -> one b128
#pragma unroll
        for (int i = 0; i < 2; ++i)
#pragma unroll
            for (int j = 0; j < 2; ++j) {
                int c  = nBase + n0 + j * 16 + laneLo;
                int mg = mBase + m0 + i * 16 + sel * 8;
                int b  = mg >> 11;            // / T_
                int t  = mg & (T_ - 1);
                union { _Float16 h[8]; float4 f4; } pk;
#pragma unroll
                for (int r = 0; r < 8; ++r) pk.h[r] = (_Float16)acc[i][j][r];
                *(float4*)(Vt + ((size_t)b * C_ + c) * T_ + t) = pk.f4;
            }
    } else {
        _Float16* out = (which == 0) ? Q : K;
#pragma unroll
        for (int i = 0; i < 2; ++i)
#pragma unroll
            for (int j = 0; j < 2; ++j) {
                int n = nBase + n0 + j * 16 + laneLo;
#pragma unroll
                for (int r = 0; r < 8; ++r) {
                    int m = mBase + m0 + i * 16 + sel * 8 + r;
                    out[(size_t)m * C_ + n] = (_Float16)acc[i][j][r];   // coalesced across lanes
                }
            }
    }
}

// ---- kernel 2: S = (Q K^T) / sqrt(C), causal tiles only ------------------
__global__ __launch_bounds__(256) void scores_kernel(
    const _Float16* __restrict__ Q, const _Float16* __restrict__ Km, _Float16* __restrict__ S)
{
    const int b = blockIdx.z;
    const int mBase = blockIdx.y * TM;
    const int nBase = blockIdx.x * TN;
    if (nBase > mBase + TM - 1) return;   // fully masked tile

    const _Float16* Qb = Q  + (size_t)b * T_ * C_;
    const _Float16* Kb = Km + (size_t)b * T_ * C_;
    _Float16*       Sb = S  + (size_t)b * T_ * T_;

    __shared__ __align__(16) _Float16 sA[TM * LDSS];
    __shared__ __align__(16) _Float16 sB[TN * LDSS];

    const int tid = threadIdx.x;
    const int wave = tid >> 5, lane = tid & 31, laneLo = lane & 15, sel = lane >> 4;
    const int m0 = (wave >> 1) * 32, n0 = (wave & 1) * 32;

    const v8f vz = {0, 0, 0, 0, 0, 0, 0, 0};
    v8f acc[2][2] = {{vz, vz}, {vz, vz}};

    for (int kk = 0; kk < C_; kk += TK) {
        __syncthreads();
#pragma unroll
        for (int p = 0; p < 2; ++p) {
            int row = p * 64 + (tid >> 2);
            int col = (tid & 3) * 8;
            *(v8h*)(sA + row * LDSS + col) =
                *(const v8h*)(Qb + (size_t)(mBase + row) * C_ + kk + col);
        }
        {
            int row = tid >> 2;
            int col = (tid & 3) * 8;
            *(v8h*)(sB + row * LDSS + col) =
                *(const v8h*)(Kb + (size_t)(nBase + row) * C_ + kk + col);
        }
        __syncthreads();
        wave_mma_step(sA, sB, m0, n0, laneLo, sel, acc);
    }

    const float scale = 0.03125f;  // 1/sqrt(1024)
#pragma unroll
    for (int i = 0; i < 2; ++i)
#pragma unroll
        for (int j = 0; j < 2; ++j) {
            int n = nBase + n0 + j * 16 + laneLo;
#pragma unroll
            for (int r = 0; r < 8; ++r) {
                int m = mBase + m0 + i * 16 + sel * 8 + r;
                Sb[(size_t)m * T_ + n] = (_Float16)(acc[i][j][r] * scale);
            }
        }
}

// ---- kernel 3: in-place masked row softmax S -> P ------------------------
__global__ __launch_bounds__(256) void softmax_kernel(_Float16* __restrict__ S)
{
    const int row = blockIdx.x;            // 0 .. B*T-1
    const int m = row & (T_ - 1);
    _Float16* p = S + (size_t)row * T_;
    const int L = m + 1;                   // valid length (causal)
    const int tid = threadIdx.x;

    __shared__ float red[256];
    float v[T_ / 256];
    float mx = -3.0e38f;
#pragma unroll
    for (int i = 0; i < T_ / 256; ++i) {
        int n = tid + i * 256;
        float s = (n < L) ? (float)p[n] : -3.0e38f;
        v[i] = s;
        mx = fmaxf(mx, s);
    }
    red[tid] = mx; __syncthreads();
    for (int off = 128; off > 0; off >>= 1) {
        if (tid < off) red[tid] = fmaxf(red[tid], red[tid + off]);
        __syncthreads();
    }
    mx = red[0]; __syncthreads();

    float sum = 0.0f;
#pragma unroll
    for (int i = 0; i < T_ / 256; ++i) {
        int n = tid + i * 256;
        float e = (n < L) ? __expf(v[i] - mx) : 0.0f;
        v[i] = e; sum += e;
    }
    red[tid] = sum; __syncthreads();
    for (int off = 128; off > 0; off >>= 1) {
        if (tid < off) red[tid] += red[tid + off];
        __syncthreads();
    }
    const float inv = 1.0f / red[0];
#pragma unroll
    for (int i = 0; i < T_ / 256; ++i) {
        int n = tid + i * 256;
        p[n] = (_Float16)((n < L) ? v[i] * inv : 0.0f);   // zeros past diagonal
    }
}

// ---- kernel 4: O = P @ V  (V consumed as Vt rows) ------------------------
__global__ __launch_bounds__(256) void out_kernel(
    const _Float16* __restrict__ P, const _Float16* __restrict__ Vt, float* __restrict__ O)
{
    const int b = blockIdx.z;
    const int mBase = blockIdx.y * TM;
    const int nBase = blockIdx.x * TN;
    const _Float16* Pb = P  + (size_t)b * T_ * T_;
    const _Float16* Vb = Vt + (size_t)b * C_ * T_;
    float*          Ob = O  + (size_t)b * T_ * C_;

    __shared__ __align__(16) _Float16 sA[TM * LDSS];
    __shared__ __align__(16) _Float16 sB[TN * LDSS];

    const int tid = threadIdx.x;
    const int wave = tid >> 5, lane = tid & 31, laneLo = lane & 15, sel = lane >> 4;
    const int m0 = (wave >> 1) * 32, n0 = (wave & 1) * 32;

    const v8f vz = {0, 0, 0, 0, 0, 0, 0, 0};
    v8f acc[2][2] = {{vz, vz}, {vz, vz}};

    const int kEnd = mBase + TM;   // causal: P[m][n]==0 for n > m

    for (int kk = 0; kk < kEnd; kk += TK) {
        __syncthreads();
#pragma unroll
        for (int p = 0; p < 2; ++p) {
            int row = p * 64 + (tid >> 2);
            int col = (tid & 3) * 8;
            *(v8h*)(sA + row * LDSS + col) =
                *(const v8h*)(Pb + (size_t)(mBase + row) * T_ + kk + col);
        }
        {
            int row = tid >> 2;
            int col = (tid & 3) * 8;
            *(v8h*)(sB + row * LDSS + col) =
                *(const v8h*)(Vb + (size_t)(nBase + row) * T_ + kk + col);
        }
        __syncthreads();
        wave_mma_step(sA, sB, m0, n0, laneLo, sel, acc);
    }

#pragma unroll
    for (int i = 0; i < 2; ++i)
#pragma unroll
        for (int j = 0; j < 2; ++j) {
            int n = nBase + n0 + j * 16 + laneLo;
#pragma unroll
            for (int r = 0; r < 8; ++r) {
                int m = mBase + m0 + i * 16 + sel * 8 + r;
                Ob[(size_t)m * C_ + n] = acc[i][j][r];
            }
        }
}

// ---- host launcher -------------------------------------------------------
extern "C" void kernel_launch(void* const* d_in, const int* in_sizes, int n_in,
                              void* d_out, int out_size, void* d_ws, size_t ws_size,
                              hipStream_t stream) {
    (void)in_sizes; (void)n_in; (void)out_size; (void)ws_size;
    const float* x  = (const float*)d_in[0];
    const float* Wk = (const float*)d_in[1];   // setup_inputs order: x, Wk, Wq, Wv
    const float* Wq = (const float*)d_in[2];
    const float* Wv = (const float*)d_in[3];
    float* out = (float*)d_out;

    char* ws = (char*)d_ws;
    _Float16* Q  = (_Float16*)(ws);                              // 16 MB
    _Float16* K  = (_Float16*)(ws + (size_t)BT_ * C_ * 2);       // 16 MB
    _Float16* Vt = (_Float16*)(ws + (size_t)BT_ * C_ * 4);       // 16 MB, [B][C][T]
    _Float16* S  = (_Float16*)(ws + (size_t)BT_ * C_ * 6);       // 32 MB, [B][T][T]

    qkv_kernel   <<<dim3(C_ / TN, BT_ / TM, 3), 256, 0, stream>>>(x, Wq, Wk, Wv, Q, K, Vt);
    scores_kernel<<<dim3(T_ / TN, T_ / TM, B_), 256, 0, stream>>>(Q, K, S);
    softmax_kernel<<<dim3(BT_), 256, 0, stream>>>(S);
    out_kernel   <<<dim3(C_ / TN, T_ / TM, B_), 256, 0, stream>>>(S, Vt, out);
}
